// DynamicMetaEmbedding_48996986913208
// MI455X (gfx1250) — matrix operand order repaired and checked by reading, not compiled, not run on hardware
//
#include <hip/hip_runtime.h>

// ---------------- problem constants ----------------
#define D_OUT 256
#define K0 300
#define K1 300
#define K2 200
#define KP0 320                     // each region padded to multiple of 32
#define KP1 320
#define KP2 224
#define KPT (KP0 + KP1 + KP2)       // 864 = stacked, padded K
#define NKT (KPT / 32)              // 27 WMMA K-steps
#define TOK_PER_WG 64
#define NTOK (32 * 512)             // 16384 tokens

// d_ws layout: packed bf16 W fragments, then v = concat(W_n @ Wa) in f32
#define WS_W_ELEMS (NKT * 16 * 512)       // 221184 bf16
#define WS_V_BYTEOFF (WS_W_ELEMS * 2)     // 442368 B (16B aligned)

typedef __attribute__((ext_vector_type(16))) __bf16          v16bf;
typedef __attribute__((ext_vector_type(8)))  __bf16          bf16x8;
typedef __attribute__((ext_vector_type(8)))  float           v8f;
typedef __attribute__((ext_vector_type(4)))  float           f32x4;
typedef __attribute__((ext_vector_type(4)))  unsigned int    u32x4;
typedef __attribute__((ext_vector_type(4)))  unsigned short  u16x4;

union Frag { u32x4 u[2]; v16bf v; };

__device__ __forceinline__ unsigned short f2bf(float f) {
  unsigned int u = __builtin_bit_cast(unsigned int, f);
  u += 0x7FFFu + ((u >> 16) & 1u);          // round-to-nearest-even
  return (unsigned short)(u >> 16);
}

// ---------------------------------------------------------------------------
// Kernel 0: pack [W0;W1;W2] (f32, [K][256]) -> bf16 WMMA 32x16 B-fragments in
// d_ws, indexed by the stacked padded K (kp in [0,864)).
// block(kt,nt) = 512 bf16: lane = (n&15) + 16*((kp>>4)&1), elem = kp&15.
// ---------------------------------------------------------------------------
__global__ void pack_w_kernel(const float* __restrict__ W0,
                              const float* __restrict__ W1,
                              const float* __restrict__ W2,
                              unsigned short* __restrict__ ws) {
  int i   = blockIdx.x * blockDim.x + threadIdx.x;  // 0 .. 864*256-1
  int n   = i & 255;
  int row = i >> 8;                                 // stacked padded-k, 0..863
  const float* W;
  int k, Ks;
  if (row < KP0)            { W = W0; k = row;             Ks = K0; }
  else if (row < KP0 + KP1) { W = W1; k = row - KP0;       Ks = K1; }
  else                      { W = W2; k = row - KP0 - KP1; Ks = K2; }

  float val = (k < Ks) ? W[k * D_OUT + n] : 0.0f;
  int kt   = row >> 5;
  int half = (row >> 4) & 1;
  int e    = row & 15;
  int nt   = n >> 4;
  int lane = (n & 15) + (half << 4);
  ws[(kt * 16 + nt) * 512 + lane * 16 + e] = f2bf(val);
}

// ---------------------------------------------------------------------------
// Kernel 1: v[kp] = (W_s @ Wa)[k]  (f32, 864 values; zero in padded rows).
// One block per kp, 256 threads, coalesced row read + wave/LDS reduction.
// ---------------------------------------------------------------------------
__global__ void compute_v_kernel(const float* __restrict__ W0,
                                 const float* __restrict__ W1,
                                 const float* __restrict__ W2,
                                 const float* __restrict__ Wa,
                                 float* __restrict__ v) {
  const int kp = blockIdx.x;
  const int t  = threadIdx.x;
  const float* W;
  int k, Ks;
  if (kp < KP0)            { W = W0; k = kp;             Ks = K0; }
  else if (kp < KP0 + KP1) { W = W1; k = kp - KP0;       Ks = K1; }
  else                     { W = W2; k = kp - KP0 - KP1; Ks = K2; }

  float p = (k < Ks) ? W[k * D_OUT + t] * Wa[t] : 0.0f;
  p += __shfl_xor(p, 1);
  p += __shfl_xor(p, 2);
  p += __shfl_xor(p, 4);
  p += __shfl_xor(p, 8);
  p += __shfl_xor(p, 16);
  __shared__ float sp[8];
  if ((t & 31) == 0) sp[t >> 5] = p;
  __syncthreads();
  if (t == 0) {
    float s = 0.0f;
#pragma unroll
    for (int i = 0; i < 8; ++i) s += sp[i];
    v[kp] = s;
  }
}

// ---------------------------------------------------------------------------
// Gather one region into LDS bf16 (zero-padded) and return the fused partial
// logit dot  sum_k x[k]*v[k]  for this sub-thread. 4 sub-threads per token.
// ---------------------------------------------------------------------------
__device__ __forceinline__ float gather_dot(const float* __restrict__ src,
                                            const float* __restrict__ vv,
                                            unsigned short* dst,
                                            int st, int Kv, int Kp) {
  float acc = 0.0f;
  for (int k4 = st * 4; k4 < Kp; k4 += 16) {
    u16x4 pk;
    if (k4 < Kv) {
      f32x4 x = *reinterpret_cast<const f32x4*>(src + k4);
      f32x4 c = *reinterpret_cast<const f32x4*>(vv + k4);
      acc += x.x * c.x + x.y * c.y + x.z * c.z + x.w * c.w;
      pk.x = f2bf(x.x); pk.y = f2bf(x.y); pk.z = f2bf(x.z); pk.w = f2bf(x.w);
    } else {
      pk.x = 0; pk.y = 0; pk.z = 0; pk.w = 0;
    }
    *reinterpret_cast<u16x4*>(dst + k4) = pk;
  }
  return acc;
}

// Scale one bf16 region of a token row in LDS by alpha (8 elems / iteration).
__device__ __forceinline__ void scale_region(unsigned short* p, int st, int Kp,
                                             float a) {
  const __bf16 ab = (__bf16)a;
  for (int k8 = st * 8; k8 < Kp; k8 += 32) {
    bf16x8 x = *reinterpret_cast<bf16x8*>(p + k8);
    x = x * ab;                       // v_pk_mul_bf16 / f32-promoted VALU
    *reinterpret_cast<bf16x8*>(p + k8) = x;
  }
}

// ---------------------------------------------------------------------------
// Main kernel: 256 threads (8 wave32), 64 tokens per WG.
//  1. gather 3 embedding rows -> LDS bf16, fused logit dot x.(W@Wa)
//  2. 3-way softmax per token (ba cancels, b* are zero)
//  3. scale LDS rows by alpha  =>  out = [a0*x0,a1*x1,a2*x2] @ [W0;W1;W2]
//  4. single stacked K=864 bf16 WMMA GEMM, 4 M-tiles x 2 N-tiles per wave
// ---------------------------------------------------------------------------
__global__ void __launch_bounds__(256, 1)
meta_embed_kernel(const int* __restrict__ ids,
                  const float* __restrict__ E0,
                  const float* __restrict__ E1,
                  const float* __restrict__ E2,
                  const unsigned short* __restrict__ wsW,
                  const float* __restrict__ v,
                  float* __restrict__ out) {
  __shared__ __align__(16) unsigned short shA[TOK_PER_WG * KPT];  // 108 KB
  __shared__ float shL[3][TOK_PER_WG];

  const int tid = threadIdx.x;
  const int m0  = blockIdx.x * TOK_PER_WG;

  // ---- stage 1: gather + fused logit partials (4 sub-threads per token) ----
  {
    const int m  = tid >> 2;
    const int st = tid & 3;
    const long long id = (long long)ids[m0 + m];
    unsigned short* arow = &shA[m * KPT];
    float l0 = gather_dot(E0 + id * K0, v,             arow,             st, K0, KP0);
    float l1 = gather_dot(E1 + id * K1, v + KP0,       arow + KP0,       st, K1, KP1);
    float l2 = gather_dot(E2 + id * K2, v + KP0 + KP1, arow + KP0 + KP1, st, K2, KP2);
    l0 += __shfl_xor(l0, 1, 4); l0 += __shfl_xor(l0, 2, 4);
    l1 += __shfl_xor(l1, 1, 4); l1 += __shfl_xor(l1, 2, 4);
    l2 += __shfl_xor(l2, 1, 4); l2 += __shfl_xor(l2, 2, 4);
    if (st == 0) { shL[0][m] = l0; shL[1][m] = l1; shL[2][m] = l2; }
  }
  __syncthreads();

  // ---- stage 2: softmax over the 3 sources ----
  if (tid < TOK_PER_WG) {
    float l0 = shL[0][tid], l1 = shL[1][tid], l2 = shL[2][tid];
    float mx = fmaxf(l0, fmaxf(l1, l2));
    float e0 = __expf(l0 - mx), e1 = __expf(l1 - mx), e2 = __expf(l2 - mx);
    float inv = 1.0f / (e0 + e1 + e2);
    shL[0][tid] = e0 * inv;
    shL[1][tid] = e1 * inv;
    shL[2][tid] = e2 * inv;
  }
  __syncthreads();

  // ---- stage 3: scale LDS rows by alpha (per source region) ----
  {
    const int m  = tid >> 2;
    const int st = tid & 3;
    unsigned short* arow = &shA[m * KPT];
    scale_region(arow,             st, KP0, shL[0][m]);
    scale_region(arow + KP0,       st, KP1, shL[1][m]);
    scale_region(arow + KP0 + KP1, st, KP2, shL[2][m]);
  }
  __syncthreads();

  // ---- stage 4: one stacked GEMM: [64 x 864] @ [864 x 256] ----
  const int wave = tid >> 5;
  const int lane = tid & 31;
  const int half = lane >> 4;
  const int mrow = lane & 15;
  const int nt0  = wave * 2;

  v8f acc[4][2];
#pragma unroll
  for (int mt = 0; mt < 4; ++mt)
#pragma unroll
    for (int j = 0; j < 2; ++j)
#pragma unroll
      for (int r = 0; r < 8; ++r) acc[mt][j][r] = 0.0f;

#pragma unroll
  for (int kt = 0; kt < NKT; ++kt) {
    // B fragments (bf16, pre-swizzled, contiguous 1KB per wave, L2-resident)
    const unsigned short* wb = wsW + (kt * 16 + nt0) * 512 + lane * 16;
    Frag b0, b1;
    b0.u[0] = *reinterpret_cast<const u32x4*>(wb);
    b0.u[1] = *reinterpret_cast<const u32x4*>(wb + 8);
    b1.u[0] = *reinterpret_cast<const u32x4*>(wb + 512);
    b1.u[1] = *reinterpret_cast<const u32x4*>(wb + 520);
#pragma unroll
    for (int mt = 0; mt < 4; ++mt) {
      // A fragment from LDS: contiguous K layout (lane<16: K{0..15}, lane>=16: K{16..31})
      const unsigned short* ab =
          &shA[(mt * 16 + mrow) * KPT + kt * 32 + half * 16];
      Frag a;
      a.u[0] = *reinterpret_cast<const u32x4*>(ab);
      a.u[1] = *reinterpret_cast<const u32x4*>(ab + 8);
      acc[mt][0] = __builtin_amdgcn_wmma_f32_16x16x32_bf16(
          false, a.v, false, b0.v, (short)0, acc[mt][0], false, false);
      acc[mt][1] = __builtin_amdgcn_wmma_f32_16x16x32_bf16(
          false, a.v, false, b1.v, (short)0, acc[mt][1], false, false);
    }
  }

  // ---- stage 5: direct store (C layout: VGPR r -> M = half*8 + r) ----
#pragma unroll
  for (int mt = 0; mt < 4; ++mt)
#pragma unroll
    for (int r = 0; r < 8; ++r) {
      const int m = mt * 16 + half * 8 + r;
      float* op = out + (long long)(m0 + m) * D_OUT + mrow;
      op[nt0 * 16]      = acc[mt][0][r];
      op[nt0 * 16 + 16] = acc[mt][1][r];
    }
}

// ---------------------------------------------------------------------------
extern "C" void kernel_launch(void* const* d_in, const int* in_sizes, int n_in,
                              void* d_out, int out_size, void* d_ws, size_t ws_size,
                              hipStream_t stream) {
  (void)in_sizes; (void)n_in; (void)out_size; (void)ws_size;
  const int*   ids = (const int*)  d_in[0];
  const float* E0  = (const float*)d_in[1];
  const float* E1  = (const float*)d_in[2];
  const float* E2  = (const float*)d_in[3];
  const float* W0  = (const float*)d_in[4];
  const float* W1  = (const float*)d_in[6];
  const float* W2  = (const float*)d_in[8];
  const float* Wa  = (const float*)d_in[10];
  unsigned short* wsW = (unsigned short*)d_ws;
  float*          wsV = (float*)((char*)d_ws + WS_V_BYTEOFF);
  float* out = (float*)d_out;

  // Kernel 0: pre-swizzle stacked W into bf16 WMMA B-fragment layout.
  pack_w_kernel<<<WS_W_ELEMS / 256, 256, 0, stream>>>(W0, W1, W2, wsW);

  // Kernel 1: v = concat(W_n @ Wa)  (folds attention Dense into a vector).
  compute_v_kernel<<<KPT, 256, 0, stream>>>(W0, W1, W2, Wa, wsV);

  // Kernel 2: gather + fused logits + softmax + alpha-scale + stacked GEMM.
  meta_embed_kernel<<<NTOK / TOK_PER_WG, 256, 0, stream>>>(ids, E0, E1, E2,
                                                           wsW, wsV, out);
}
